// Decoder_40321152975658
// MI455X (gfx1250) — matrix operand order repaired and checked by reading, not compiled, not run on hardware
//
#include <hip/hip_runtime.h>
#include <hip/hip_bf16.h>

// ---------------------------------------------------------------------------
// Decoder (LSTM + degenerate Luong attention) for MI455X / gfx1250.
//
//  * softmax over size-1 axis => ctx[b] = sum_s enc[b,s,:]; attn_W dead.
//  * x_t @ W_ih^T hoisted into one big GEMM G_x; ctx part of comb hoisted.
//  * logits GEMM ([2048x1024]x[1024x32000], ~134 GFLOP) dominates; weights
//    pre-converted to bf16 (L2-resident) -> v_wmma_f32_16x16x32_bf16.
//  * B-tile staged with gfx1250 async copy: global_load_async_to_lds_b128
//    + s_wait_asynccnt (ASYNCcnt path), A-tile converted f32->bf16 in VALU.
// ---------------------------------------------------------------------------

typedef __attribute__((ext_vector_type(16))) __bf16 v16bf;
typedef __attribute__((ext_vector_type(8)))  float  v8f;

#define B_  64
#define T_  32
#define S_  64
#define E_  1024
#define H_  1024
#define V_  32000

#define BM 64
#define BN 256
#define BK 64
#define LDT 72   // LDS row stride in ushorts (144 B: 16B-aligned, bank-skewed)

#define USE_ASYNC_LDS 1

__device__ __forceinline__ unsigned pack_bf16(float x, float y) {
    union { __bf16 h[2]; unsigned u; } p;
    p.h[0] = (__bf16)x;              // native packed cvt on gfx1250, RNE
    p.h[1] = (__bf16)y;
    return p.u;
}
__device__ __forceinline__ unsigned short f2bf(float x) {
    union { __bf16 h; unsigned short s; } p;
    p.h = (__bf16)x;
    return p.s;
}
__device__ __forceinline__ float sigm(float x) { return 1.0f / (1.0f + __expf(-x)); }

#if USE_ASYNC_LDS
// Async global->LDS 16B copy (gfx1250): vdst = LDS byte address, vaddr = 64-bit
// global address. Tracked by ASYNCcnt.
__device__ __forceinline__ void async_ld_b128(unsigned lds, const void* gptr) {
    asm volatile("global_load_async_to_lds_b128 %0, %1, off"
                 :: "v"(lds), "v"(gptr) : "memory");
}
__device__ __forceinline__ void wait_asynccnt0() {
    asm volatile("s_wait_asynccnt 0x0" ::: "memory");
}
#endif

union FragBF { v16bf v; uint4 q[2]; };

// A fragment (16x32 bf16, ISA 7.12.2): lane L -> row = L%16;
// lanes 0-15 hold K = {0..7, 16..23}, lanes 16-31 hold K = {8..15, 24..31}.
__device__ __forceinline__ v16bf load_a_frag(const unsigned short* sm, int mo, int kk, int lane) {
    int row  = mo + (lane & 15);
    int kb   = kk * 32 + (lane >> 4) * 8;
    FragBF f;
    f.q[0] = *(const uint4*)(sm + row * LDT + kb);
    f.q[1] = *(const uint4*)(sm + row * LDT + kb + 16);
    return f.v;
}
// B fragment (32x16 bf16): lane L -> col = L%16;
// lanes 0-15 hold K = {0..7, 16..23}, lanes 16-31 hold K = {8..15, 24..31} (same as A).
__device__ __forceinline__ v16bf load_b_frag(const unsigned short* sm, int no, int kk, int lane) {
    int col  = no + (lane & 15);
    int kb   = kk * 32 + (lane >> 4) * 8;
    FragBF f;
    f.q[0] = *(const uint4*)(sm + col * LDT + kb);
    f.q[1] = *(const uint4*)(sm + col * LDT + kb + 16);
    return f.v;
}

// C = act( A_f32[M,K] @ W_bf16[N,K]^T + bias[N]? + C0[M,N]? )
// Requires M%64==0, N%256==0, K%64==0 (all call sites satisfy this).
__global__ __launch_bounds__(256)
void k_gemm_bf16(const float* __restrict__ A, int lda,
                 const unsigned short* __restrict__ W, int ldw,
                 const float* __restrict__ bias,
                 const float* __restrict__ C0, int ldc0,
                 float* __restrict__ C, int ldc,
                 int M, int N, int K, int act) {
    __shared__ unsigned short smA[BM * LDT];
    __shared__ unsigned short smB[BN * LDT];

    const int tid  = threadIdx.x;
    const int lane = tid & 31;
    const int w    = tid >> 5;          // 8 waves
    const int wm   = (w & 1) * 32;      // 2 waves along M (32 rows each)
    const int wn   = (w >> 1) * 64;     // 4 waves along N (64 cols each)
    const int bM   = blockIdx.y * BM;
    const int bN   = blockIdx.x * BN;

    v8f acc[2][4] = {};

    for (int k0 = 0; k0 < K; k0 += BK) {
        __syncthreads();
        // Stage A (64x64 f32 -> bf16): 4 float4 per thread.
        #pragma unroll
        for (int it = 0; it < 4; ++it) {
            int idx = tid + it * 256;           // float4 index
            int row = idx >> 4;
            int c4  = (idx & 15) << 2;
            const float4 v = *(const float4*)(A + (size_t)(bM + row) * lda + k0 + c4);
            uint2 p;
            p.x = pack_bf16(v.x, v.y);
            p.y = pack_bf16(v.z, v.w);
            *(uint2*)(smA + row * LDT + c4) = p;
        }
        // Stage B (256x64 bf16): 8 x 16B per thread, async global->LDS.
        #pragma unroll
        for (int it = 0; it < 8; ++it) {
            int idx = tid + it * 256;           // uint4 index (8 bf16)
            int row = idx >> 3;
            int c8  = (idx & 7) << 3;
            const unsigned short* g = W + (size_t)(bN + row) * ldw + k0 + c8;
#if USE_ASYNC_LDS
            async_ld_b128((unsigned)(uintptr_t)(smB + row * LDT + c8), g);
#else
            *(uint4*)(smB + row * LDT + c8) = *(const uint4*)g;
#endif
        }
        if (k0 + BK < K)   // prefetch next A slab (global_prefetch_b8)
            __builtin_prefetch(A + (size_t)(bM + (tid >> 2)) * lda + k0 + BK, 0, 3);
#if USE_ASYNC_LDS
        wait_asynccnt0();                       // our async LDS writes done
#endif
        __syncthreads();

        #pragma unroll
        for (int kk = 0; kk < 2; ++kk) {
            v16bf a0 = load_a_frag(smA, wm,      kk, lane);
            v16bf a1 = load_a_frag(smA, wm + 16, kk, lane);
            #pragma unroll
            for (int tn = 0; tn < 4; ++tn) {
                v16bf b = load_b_frag(smB, wn + tn * 16, kk, lane);
                acc[0][tn] = __builtin_amdgcn_wmma_f32_16x16x32_bf16(false, a0, false, b, (short)0, acc[0][tn], false, false);
                acc[1][tn] = __builtin_amdgcn_wmma_f32_16x16x32_bf16(false, a1, false, b, (short)0, acc[1][tn], false, false);
            }
        }
    }

    // Epilogue. C/D layout: VGPR r -> lanes 0-15: M=r, lanes 16-31: M=8+r; N=lane%16.
    const int half = lane >> 4;
    const int ln   = lane & 15;
    #pragma unroll
    for (int tm = 0; tm < 2; ++tm) {
        #pragma unroll
        for (int tn = 0; tn < 4; ++tn) {
            #pragma unroll
            for (int r = 0; r < 8; ++r) {
                int row = bM + wm + tm * 16 + half * 8 + r;
                int col = bN + wn + tn * 16 + ln;
                float v = acc[tm][tn][r];
                if (bias) v += bias[col];
                if (C0)   v += C0[(size_t)row * ldc0 + col];
                if (act)  v = tanhf(v);
                C[(size_t)row * ldc + col] = v;
            }
        }
    }
}

// dst[r*K + k] = bf16(src[r*ld + off + k])   (weight pre-conversion, strided)
__global__ void k_convert_bf16(const float* __restrict__ src, unsigned short* __restrict__ dst,
                               int K, int ld, int off, long long total) {
    long long i = (long long)blockIdx.x * blockDim.x + threadIdx.x;
    if (i >= total) return;
    long long r = i / K;
    int k = (int)(i - r * K);
    dst[i] = f2bf(src[r * (long long)ld + off + k]);
}

// Xemb[r][e] = emb[inputs[r]][e]
__global__ void k_gather_emb(const int* __restrict__ idx, const float* __restrict__ emb,
                             float* __restrict__ xemb) {
    int i = blockIdx.x * blockDim.x + threadIdx.x;   // 0 .. 2048*1024-1
    int r = i >> 10, e = i & 1023;
    xemb[i] = emb[(size_t)idx[r] * E_ + e];
}

// ctx[b][d] = sum_s enc[b][s][d]   (degenerate softmax => plain sum)
__global__ void k_ctx_reduce(const float* __restrict__ enc, float* __restrict__ ctx) {
    int i = blockIdx.x * blockDim.x + threadIdx.x;   // 0 .. 64*2048-1
    int b = i >> 11, d = i & 2047;
    const float* p = enc + (size_t)b * S_ * (2 * H_) + d;
    float s = 0.f;
    #pragma unroll 4
    for (int j = 0; j < S_; ++j) s += p[(size_t)j * (2 * H_)];
    ctx[i] = s;
}

__global__ void k_bias_sum(const float* __restrict__ a, const float* __restrict__ b,
                           float* __restrict__ o) {
    int i = blockIdx.x * blockDim.x + threadIdx.x;
    if (i < 4 * H_) o[i] = a[i] + b[i];
}

// PyTorch gate order i,f,g,o; updates h,c in place.
__global__ void k_lstm_cell(const float* __restrict__ gates, float* __restrict__ h,
                            float* __restrict__ c) {
    int i = blockIdx.x * blockDim.x + threadIdx.x;   // 0 .. 64*1024-1
    int b = i >> 10, j = i & 1023;
    const float* g = gates + (size_t)b * (4 * H_);
    float ig = sigm(g[j]);
    float fg = sigm(g[H_ + j]);
    float gg = tanhf(g[2 * H_ + j]);
    float og = sigm(g[3 * H_ + j]);
    float cn = fg * c[i] + ig * gg;
    c[i] = cn;
    h[i] = og * tanhf(cn);
}

extern "C" void kernel_launch(void* const* d_in, const int* in_sizes, int n_in,
                              void* d_out, int out_size, void* d_ws, size_t ws_size,
                              hipStream_t stream) {
    const int*   inputs = (const int*)  d_in[0];
    const float* enc    = (const float*)d_in[1];
    const float* h0     = (const float*)d_in[2];
    const float* c0     = (const float*)d_in[3];
    const float* emb    = (const float*)d_in[4];
    const float* W_ih   = (const float*)d_in[5];
    const float* W_hh   = (const float*)d_in[6];
    const float* b_ih   = (const float*)d_in[7];
    const float* b_hh   = (const float*)d_in[8];
    /* d_in[9] = attn_W: dead code (softmax over size-1 axis) */
    const float* comb_W = (const float*)d_in[10];
    const float* comb_b = (const float*)d_in[11];
    const float* out_W  = (const float*)d_in[12];
    const float* out_b  = (const float*)d_in[13];
    float* out = (float*)d_out;

    // ---- workspace carving (256B aligned) ----
    char* base = (char*)d_ws;
    size_t off = 0;
    auto carve = [&](size_t bytes) { char* p = base + off; off = (off + bytes + 255) & ~(size_t)255; return p; };
    unsigned short* wWih   = (unsigned short*)carve((size_t)4 * H_ * E_ * 2);
    unsigned short* wWhh   = (unsigned short*)carve((size_t)4 * H_ * H_ * 2);
    unsigned short* wCombH = (unsigned short*)carve((size_t)H_ * H_ * 2);
    unsigned short* wCombC = (unsigned short*)carve((size_t)H_ * 2 * H_ * 2);
    unsigned short* wOutW  = (unsigned short*)carve((size_t)V_ * H_ * 2);
    float* xemb  = (float*)carve((size_t)B_ * T_ * E_ * 4);
    float* ctx   = (float*)carve((size_t)B_ * 2 * H_ * 4);
    float* ctxp  = (float*)carve((size_t)B_ * H_ * 4);
    float* gx    = (float*)carve((size_t)B_ * T_ * 4 * H_ * 4);
    float* gates = (float*)carve((size_t)B_ * 4 * H_ * 4);
    float* hbuf  = (float*)carve((size_t)B_ * H_ * 4);
    float* cbuf  = (float*)carve((size_t)B_ * H_ * 4);
    float* outs  = (float*)carve((size_t)B_ * T_ * H_ * 4);
    float* bsum  = (float*)carve((size_t)4 * H_ * 4);

    // ---- 1. weight pre-conversion f32 -> bf16 ----
    auto cvt = [&](const float* s, unsigned short* d, int K, int ld, int o, long long tot) {
        k_convert_bf16<<<(unsigned)((tot + 255) / 256), 256, 0, stream>>>(s, d, K, ld, o, tot);
    };
    cvt(W_ih,  wWih,   E_,     E_,     0,  (long long)4 * H_ * E_);
    cvt(W_hh,  wWhh,   H_,     H_,     0,  (long long)4 * H_ * H_);
    cvt(comb_W, wCombH, H_,    3 * H_, 0,  (long long)H_ * H_);
    cvt(comb_W, wCombC, 2 * H_, 3 * H_, H_, (long long)H_ * 2 * H_);
    cvt(out_W, wOutW,  H_,     H_,     0,  (long long)V_ * H_);

    // ---- 2. embedding gather, ctx reduction, bias sum ----
    k_gather_emb<<<(B_ * T_ * E_) / 256, 256, 0, stream>>>(inputs, emb, xemb);
    k_ctx_reduce<<<(B_ * 2 * H_) / 256, 256, 0, stream>>>(enc, ctx);
    k_bias_sum<<<(4 * H_) / 256, 256, 0, stream>>>(b_ih, b_hh, bsum);

    // ---- 3. time-parallel gate input: G_x = Xemb @ W_ih^T + (b_ih + b_hh) ----
    k_gemm_bf16<<<dim3((4 * H_) / BN, (B_ * T_) / BM), 256, 0, stream>>>(
        xemb, E_, wWih, E_, bsum, nullptr, 0, gx, 4 * H_, B_ * T_, 4 * H_, E_, 0);

    // ---- 4. ctx_proj = ctx @ comb_W[:,H:]^T + comb_b ----
    k_gemm_bf16<<<dim3(H_ / BN, B_ / BM), 256, 0, stream>>>(
        ctx, 2 * H_, wCombC, 2 * H_, comb_b, nullptr, 0, ctxp, H_, B_, H_, 2 * H_, 0);

    // ---- 5. init recurrent state ----
    hipMemcpyAsync(hbuf, h0, (size_t)B_ * H_ * 4, hipMemcpyDeviceToDevice, stream);
    hipMemcpyAsync(cbuf, c0, (size_t)B_ * H_ * 4, hipMemcpyDeviceToDevice, stream);

    // ---- 6. sequential scan over T ----
    for (int t = 0; t < T_; ++t) {
        // gates = G_x[:,t,:] + h @ W_hh^T
        k_gemm_bf16<<<dim3((4 * H_) / BN, B_ / BM), 256, 0, stream>>>(
            hbuf, H_, wWhh, H_, nullptr,
            gx + (size_t)t * 4 * H_, T_ * 4 * H_,
            gates, 4 * H_, B_, 4 * H_, H_, 0);
        k_lstm_cell<<<(B_ * H_) / 256, 256, 0, stream>>>(gates, hbuf, cbuf);
        // out_t = tanh(h @ comb_W[:,:H]^T + ctx_proj)  -> outs[b, t, :]
        k_gemm_bf16<<<dim3(H_ / BN, B_ / BM), 256, 0, stream>>>(
            hbuf, H_, wCombH, H_, nullptr,
            ctxp, H_,
            outs + (size_t)t * H_, T_ * H_, B_, H_, H_, 1);
    }

    // ---- 7. logits = outputs @ out_W^T + out_b  (dominant GEMM) ----
    k_gemm_bf16<<<dim3(V_ / BN, (B_ * T_) / BM), 256, 0, stream>>>(
        outs, H_, wOutW, H_, out_b, nullptr, 0, out, V_, B_ * T_, V_, H_, 0);

    // ---- 8. final h, c -> output tail ----
    float* out_h = out + (size_t)B_ * T_ * V_;
    float* out_c = out_h + (size_t)B_ * H_;
    hipMemcpyAsync(out_h, hbuf, (size_t)B_ * H_ * 4, hipMemcpyDeviceToDevice, stream);
    hipMemcpyAsync(out_c, cbuf, (size_t)B_ * H_ * 4, hipMemcpyDeviceToDevice, stream);
}
